// LearningProgressIM_21577915695524
// MI455X (gfx1250) — matrix operand order repaired and checked by reading, not compiled, not run on hardware
//
#include <hip/hip_runtime.h>
#include <math.h>

#define N_PREV  65536
#define BATCH   2048
#define DIM     128
#define NSLICES 64
#define SLICE   (N_PREV / NSLICES)   // 1024 prev rows per slice
#define QB      256                  // queries per block
#define NWAVES  16
#define TPB     (NWAVES * 32)        // 512 threads

typedef __attribute__((ext_vector_type(16))) __bf16 v16bf;
typedef __attribute__((ext_vector_type(8)))  float  v8f;

// Probe for the gfx1250 async global->LDS DMA builtin (device pass only).
#if defined(__HIP_DEVICE_COMPILE__) && __has_builtin(__builtin_amdgcn_global_load_async_to_lds_b128)
#define HAVE_ASYNC 1
#else
#define HAVE_ASYNC 0
#endif

// ---- fp32 -> bf16 (RNE) split helpers (pure integer ALU, deterministic) ----
__device__ __forceinline__ unsigned short f2bf_bits(float x) {
  unsigned u = __float_as_uint(x);
  return (unsigned short)((u + 0x7FFFu + ((u >> 16) & 1u)) >> 16);
}
__device__ __forceinline__ __bf16 bits2bf(unsigned short s) {
  return __builtin_bit_cast(__bf16, s);
}
__device__ __forceinline__ float bf_bits2f(unsigned short s) {
  return __uint_as_float(((unsigned)s) << 16);
}

// Split 4 floats into bf16 hi/lo and store packed (8B each) into LDS rows.
__device__ __forceinline__ void split_store4(__bf16* hrow, __bf16* lrow, int col, float4 x) {
  float xs0 = x.x, xs1 = x.y, xs2 = x.z, xs3 = x.w;
  unsigned short h0 = f2bf_bits(xs0), h1 = f2bf_bits(xs1);
  unsigned short h2 = f2bf_bits(xs2), h3 = f2bf_bits(xs3);
  unsigned short l0 = f2bf_bits(xs0 - bf_bits2f(h0));
  unsigned short l1 = f2bf_bits(xs1 - bf_bits2f(h1));
  unsigned short l2 = f2bf_bits(xs2 - bf_bits2f(h2));
  unsigned short l3 = f2bf_bits(xs3 - bf_bits2f(h3));
  uint2 hv, lv;
  hv.x = (unsigned)h0 | ((unsigned)h1 << 16);
  hv.y = (unsigned)h2 | ((unsigned)h3 << 16);
  lv.x = (unsigned)l0 | ((unsigned)l1 << 16);
  lv.y = (unsigned)l2 | ((unsigned)l3 << 16);
  *(uint2*)&hrow[col] = hv;
  *(uint2*)&lrow[col] = lv;
}

#if HAVE_ASYNC
// Param 0 (per hipcc diagnostic) is `v4i __device__*` i.e. a pointer to a
// 16-byte int vector in AS1; LDS side is the AS3 analogue. Cast via integers
// so we land on the exact parameter types.
typedef int v4i_t __attribute__((vector_size(16)));
typedef __attribute__((address_space(1))) v4i_t* gv4p;
typedef __attribute__((address_space(3))) v4i_t* lv4p;

__device__ __forceinline__ void async_cp16(float* lds_dst, const float* gsrc) {
  __builtin_amdgcn_global_load_async_to_lds_b128(
      (gv4p)(unsigned long long)gsrc,
      (lv4p)(unsigned)(unsigned long long)lds_dst,
      0, 0);
}
__device__ __forceinline__ void wait_async_le1() {
#if __has_builtin(__builtin_amdgcn_s_wait_asynccnt)
  __builtin_amdgcn_s_wait_asynccnt(1);
#else
  asm volatile("s_wait_asynccnt 0x1" ::: "memory");
#endif
}
__device__ __forceinline__ void wait_async_le0() {
#if __has_builtin(__builtin_amdgcn_s_wait_asynccnt)
  __builtin_amdgcn_s_wait_asynccnt(0);
#else
  asm volatile("s_wait_asynccnt 0x0" ::: "memory");
#endif
}
#endif

// ---------------- Kernel 1: ||prev_j||^2 ----------------
__global__ void lp_norms(const float* __restrict__ prev, float* __restrict__ norms) {
  const int t   = threadIdx.x;
  const int row = blockIdx.x * 32 + (t >> 2);
  const int q   = t & 3;
  const float* p = prev + (size_t)row * DIM + q * 32;
  float s = 0.f;
#pragma unroll
  for (int k = 0; k < 32; k += 4) {
    float4 v = *(const float4*)(p + k);
    s += v.x * v.x + v.y * v.y + v.z * v.z + v.w * v.w;
  }
  s += __shfl_xor(s, 1);
  s += __shfl_xor(s, 2);
  if (q == 0) norms[row] = s;
}

// ---------------- Kernel 2: fused GEMM + argmin ----------------
// grid = (BATCH/QB, NSLICES), block = 512 (16 waves).
// Wave w owns query rows [blockIdx.x*QB + 16w, +16). The block sweeps its prev
// slice in 16-row tiles; each tile is DMA'd (async, double-buffered) into LDS,
// split once into bf16 hi/lo, and consumed by all 16 waves.
// Per tile per wave: 4 K-chunks x 3 split WMMAs = 12 v_wmma_f32_16x16x32_bf16.
__global__ void __launch_bounds__(TPB)
lp_argmin(const float* __restrict__ tg, const float* __restrict__ prev,
          const float* __restrict__ norms,
          float* __restrict__ pmin, int* __restrict__ pidx) {
  __shared__ __bf16 sBh[16][DIM];          // 4 KB
  __shared__ __bf16 sBl[16][DIM];          // 4 KB
  __shared__ float  sMin[NWAVES][8][32];   // 16 KB
  __shared__ int    sIdx[NWAVES][8][32];   // 16 KB
#if HAVE_ASYNC
  __shared__ float  stageF[2][16][DIM];    // 16 KB double-buffered DMA target
#endif

  const int t    = threadIdx.x;
  const int lane = t & 31;
  const int wv   = t >> 5;
  const int half = lane >> 4;
  const int l15  = lane & 15;

  // --- A fragments: 16 queries/wave, fp32 split to bf16 hi/lo, kept in VGPRs.
  // A 16x32 bf16 layout (contiguous, matching B): lanes hold K = kb..kb+15
  // with kb = chunk + 16*half (lanes 16-31 -> +16).
  const int qrow = blockIdx.x * QB + wv * 16 + l15;
  v16bf Ahi[4], Alo[4];
  {
    const float* ar = tg + (size_t)qrow * DIM;
#pragma unroll
    for (int c = 0; c < 4; ++c) {
      const int kb = c * 32 + half * 16;
#pragma unroll
      for (int u = 0; u < 8; ++u) {
        float x = ar[kb + u];
        unsigned short hb = f2bf_bits(x);
        Ahi[c][u] = bits2bf(hb);
        Alo[c][u] = bits2bf(f2bf_bits(x - bf_bits2f(hb)));
      }
#pragma unroll
      for (int u = 0; u < 8; ++u) {
        float x = ar[kb + 8 + u];
        unsigned short hb = f2bf_bits(x);
        Ahi[c][8 + u] = bits2bf(hb);
        Alo[c][8 + u] = bits2bf(f2bf_bits(x - bf_bits2f(hb)));
      }
    }
  }

  float best[8];
  int   bestj[8];
#pragma unroll
  for (int v = 0; v < 8; ++v) { best[v] = 3.4e38f; bestj[v] = 0; }

  // Staging map: thread owns 4 floats (16 B) of the 16x128 tile.
  const int srow = t >> 5;          // 0..15
  const int scol = (t & 31) * 4;    // 0..124
  const int jbase  = blockIdx.y * SLICE;
  const int ntiles = SLICE / 16;    // 64

#if HAVE_ASYNC
  async_cp16(&stageF[0][srow][scol], prev + (size_t)(jbase + srow) * DIM + scol);
#endif

  for (int it = 0; it < ntiles; ++it) {
    const int j0 = jbase + it * 16;
#if HAVE_ASYNC
    const int cur = it & 1;
    if (it + 1 < ntiles) {
      async_cp16(&stageF[cur ^ 1][srow][scol],
                 prev + (size_t)(j0 + 16 + srow) * DIM + scol);
      wait_async_le1();             // tile `it` landed; `it+1` may be in flight
    } else {
      wait_async_le0();
    }
#endif
    __syncthreads();                // all waves done reading sB for prior tile

    float4 x;
#if HAVE_ASYNC
    x = *(const float4*)&stageF[cur][srow][scol];
#else
    if (it + 1 < ntiles)
      __builtin_prefetch(prev + (size_t)(j0 + 16 + srow) * DIM + scol, 0, 0);
    x = *(const float4*)(prev + (size_t)(j0 + srow) * DIM + scol);
#endif
    split_store4(sBh[srow], sBl[srow], scol, x);
    __syncthreads();                // sB hi/lo tile ready

    // B 32x16 bf16 layout (contiguous order, same as A): lane = column N (l15); lanes 0-15 K=0..15,
    // lanes 16-31 K=16..31 -> 16 contiguous bf16 per lane from LDS row.
    v8f acc = {};
#pragma unroll
    for (int c = 0; c < 4; ++c) {
      const int kb = c * 32 + half * 16;
      v16bf bh = *(const v16bf*)&sBh[l15][kb];
      v16bf bl = *(const v16bf*)&sBl[l15][kb];
      acc = __builtin_amdgcn_wmma_f32_16x16x32_bf16(false, Ahi[c], false, bh,
                                                    (short)0, acc, false, false);
      acc = __builtin_amdgcn_wmma_f32_16x16x32_bf16(false, Ahi[c], false, bl,
                                                    (short)0, acc, false, false);
      acc = __builtin_amdgcn_wmma_f32_16x16x32_bf16(false, Alo[c], false, bh,
                                                    (short)0, acc, false, false);
    }

    // d2 (up to a per-row constant) = ||p_j||^2 - 2 * dot
    const float nj = norms[j0 + l15];
    const int   jj = j0 + l15;
#pragma unroll
    for (int v = 0; v < 8; ++v) {
      float d = nj - 2.0f * acc[v];
      if (d < best[v]) { best[v] = d; bestj[v] = jj; }
    }
  }

  // D layout: lanes 0-15 -> rows v (0..7), lanes 16-31 -> rows 8+v; col = l15.
#pragma unroll
  for (int v = 0; v < 8; ++v) { sMin[wv][v][lane] = best[v]; sIdx[wv][v][lane] = bestj[v]; }
  __syncthreads();

  if (t < QB) {                     // one thread finalizes one of QB query rows
    const int w = t >> 4, m = t & 15;
    const int h = m >> 3, v = m & 7;
    float bm = 3.4e38f; int bi = 0;
#pragma unroll
    for (int l = 0; l < 16; ++l) {
      float x  = sMin[w][v][h * 16 + l];
      int   ix = sIdx[w][v][h * 16 + l];
      if (x < bm) { bm = x; bi = ix; }
    }
    const int q = blockIdx.x * QB + t;
    pmin[(size_t)blockIdx.y * BATCH + q] = bm;
    pidx[(size_t)blockIdx.y * BATCH + q] = bi;
  }
}

// ---------------- Kernel 3: slice-reduce + value/grad ----------------
__global__ void lp_finalize(const float* __restrict__ tg, const float* __restrict__ rg,
                            const float* __restrict__ prev,
                            const float* __restrict__ pmin, const int* __restrict__ pidx,
                            float* __restrict__ out) {
  __shared__ float sm[NSLICES];
  __shared__ int   si[NSLICES];
  __shared__ float redA[128];
  __shared__ float redB[128];
  const int i = blockIdx.x;
  const int t = threadIdx.x;

  if (t < NSLICES) {
    sm[t] = pmin[(size_t)t * BATCH + i];
    si[t] = pidx[(size_t)t * BATCH + i];
  }
  __syncthreads();
  if (t == 0) {
    float bm = sm[0]; int bi = si[0];
    for (int s = 1; s < NSLICES; ++s)
      if (sm[s] < bm) { bm = sm[s]; bi = si[s]; }
    si[0] = bi;
  }
  __syncthreads();
  const int idx = si[0];

  const float g = tg[(size_t)i * DIM + t];
  const float s = prev[(size_t)idx * DIM + t];
  const float r = rg[(size_t)i * DIM + t];
  const float ds = g - s, dr = g - r;
  redA[t] = ds * ds; redB[t] = dr * dr;
  __syncthreads();
  for (int w = 64; w > 0; w >>= 1) {
    if (t < w) { redA[t] += redA[t + w]; redB[t] += redB[t + w]; }
    __syncthreads();
  }
  const float na = sqrtf(redA[0]);
  const float nb = sqrtf(redB[0]);
  if (t == 0) out[i] = na - nb;                          // IM_vals
  out[BATCH + (size_t)i * DIM + t] = ds / na - dr / nb;  // IM_grads
}

// ---------------- launcher ----------------
extern "C" void kernel_launch(void* const* d_in, const int* in_sizes, int n_in,
                              void* d_out, int out_size, void* d_ws, size_t ws_size,
                              hipStream_t stream) {
  const float* target  = (const float*)d_in[0];
  const float* reached = (const float*)d_in[1];
  const float* tg   = target  + (size_t)N_PREV * DIM;  // last BATCH rows
  const float* rg   = reached + (size_t)N_PREV * DIM;  // last BATCH rows
  const float* prev = reached;                         // first N_PREV rows

  char* ws = (char*)d_ws;
  float* norms = (float*)ws;                                            // 256 KB
  float* pmin  = (float*)(ws + (size_t)N_PREV * 4);                     // 512 KB
  int*   pidx  = (int*)  (ws + (size_t)N_PREV * 4 +
                               (size_t)NSLICES * BATCH * 4);            // 512 KB
  float* out = (float*)d_out;

  lp_norms   <<<N_PREV / 32, 128, 0, stream>>>(prev, norms);
  lp_argmin  <<<dim3(BATCH / QB, NSLICES), TPB, 0, stream>>>(tg, prev, norms, pmin, pidx);
  lp_finalize<<<BATCH, 128, 0, stream>>>(tg, rg, prev, pmin, pidx, out);
}
